// TorchDeformConv2d_41704132444208
// MI455X (gfx1250) — compile-verified, matrix operand order checked by reading, not run on hardware
//
#include <hip/hip_runtime.h>
#include <hip/hip_bf16.h>

// ---------------------------------------------------------------------------
// Deformable conv v2 pipeline for gfx1250 (MI455X), wave32 + WMMA bf16.
// B=8, C=64, O=128, H=W=64, K=3, PAD=1.
//   Kdim = C*9 = 576,  Ncat = 2*C*9 + C*9 = 1728,  M = B*H*W = 32768.
// ---------------------------------------------------------------------------

#define Bsz   8
#define Cch   64
#define Och   128
#define Hdim  64
#define Wdim  64
#define KK    9
#define KDIM  576          // C*9
#define NOFF  1152         // 2*C*9   (offset channels)
#define NCAT  1728         // NOFF + C*9
#define MTOT  32768        // B*H*W
#define MAXOFF 16.0f       // max(H,W)/4

typedef __attribute__((ext_vector_type(16))) __bf16 v16bf;
typedef __attribute__((ext_vector_type(8)))  float  v8f;

__device__ __forceinline__ unsigned short f2bf(float f) {
  unsigned int u = __float_as_uint(f);
  u += 0x7FFFu + ((u >> 16) & 1u);          // round-to-nearest-even
  return (unsigned short)(u >> 16);
}
__device__ __forceinline__ float bf2f(unsigned short s) {
  return __uint_as_float(((unsigned int)s) << 16);
}
__device__ __forceinline__ v16bf ldfrag_g(const unsigned short* p) {
  return *reinterpret_cast<const v16bf*>(p);
}
__device__ __forceinline__ v16bf ldfrag_s(const unsigned short* p) {
  return *reinterpret_cast<const v16bf*>(p);
}
__device__ __forceinline__ v8f wmma_bf16(v16bf a, v16bf b, v8f c) {
  return __builtin_amdgcn_wmma_f32_16x16x32_bf16(
      /*neg_a=*/false, a, /*neg_b=*/false, b,
      /*c_mod=*/(short)0, c, /*reuse_a=*/false, /*reuse_b=*/false);
}

// ---------------------------------------------------------------------------
// Prep 1: bf16 im2col matrix  A[m][k],  m=(b,h,w), k=c*9+ki*3+kj
// ---------------------------------------------------------------------------
__global__ __launch_bounds__(256) void im2col_bf16(
    const float* __restrict__ x, unsigned short* __restrict__ A) {
  size_t idx = (size_t)blockIdx.x * 256 + threadIdx.x;
  if (idx >= (size_t)MTOT * KDIM) return;
  int m = (int)(idx / KDIM);
  int k = (int)(idx % KDIM);
  int c  = k / KK;
  int kk = k % KK;
  int ki = kk / 3, kj = kk % 3;
  int b = m >> 12;
  int h = (m >> 6) & 63;
  int w = m & 63;
  int hy = h + ki - 1;
  int wx = w + kj - 1;
  float v = 0.0f;
  if (hy >= 0 && hy < Hdim && wx >= 0 && wx < Wdim)
    v = x[(((size_t)b * Cch + c) << 12) + hy * Wdim + wx];
  A[idx] = f2bf(v);
}

// ---------------------------------------------------------------------------
// Prep 2: pack weights to bf16.  Wcat[n][k] = [w_off ; w_mod],  Wc[o][k]=w_conv
// ---------------------------------------------------------------------------
__global__ __launch_bounds__(256) void packw_bf16(
    const float* __restrict__ w_off, const float* __restrict__ w_mod,
    const float* __restrict__ w_conv,
    unsigned short* __restrict__ Wcat, unsigned short* __restrict__ Wc) {
  size_t idx = (size_t)blockIdx.x * 256 + threadIdx.x;
  const size_t ncat = (size_t)NCAT * KDIM;
  const size_t nc   = (size_t)Och * KDIM;
  if (idx < ncat) {
    int n = (int)(idx / KDIM);
    int k = (int)(idx % KDIM);
    float v = (n < NOFF) ? w_off[(size_t)n * KDIM + k]
                         : w_mod[(size_t)(n - NOFF) * KDIM + k];
    Wcat[idx] = f2bf(v);
  } else if (idx < ncat + nc) {
    size_t j = idx - ncat;
    Wc[j] = f2bf(w_conv[j]);
  }
}

// ---------------------------------------------------------------------------
// Main fused kernel: one 16-pixel M-tile per 256-thread block (8 wave32s).
// ---------------------------------------------------------------------------
__global__ __launch_bounds__(256) void deform_fused(
    const float* __restrict__ x,
    const float* __restrict__ b_off, const float* __restrict__ b_mod,
    const unsigned short* __restrict__ Aim,
    const unsigned short* __restrict__ Wcat,
    const unsigned short* __restrict__ Wc,
    float* __restrict__ out) {

  __shared__ __align__(32) unsigned short s_mask[16 * KDIM];  // 18.0 KB
  __shared__ __align__(32) unsigned short s_samp[16 * KDIM];  // 18.0 KB
  __shared__ __align__(32) float          s_off[8][256];      //  8.0 KB

  const int tid  = threadIdx.x;
  const int wave = tid >> 5;
  const int lane = tid & 31;
  const int col  = lane & 15;        // M/N index inside fragment
  const int khal = lane >> 4;        // which 16-wide K half
  const int m0   = blockIdx.x * 16;  // 16 pixels, same b,h; w = w0..w0+15

  const unsigned short* arow = Aim + (size_t)(m0 + col) * KDIM + khal * 16;

  // ---- Phase 1a: mask channels (N-tiles 72..107) -> s_mask ----------------
  for (int t = 72 + wave; t < 108; t += 8) {
    const int n0 = t * 16;
    v8f acc = {0.f, 0.f, 0.f, 0.f, 0.f, 0.f, 0.f, 0.f};
    const unsigned short* brow = Wcat + (size_t)(n0 + col) * KDIM + khal * 16;
    for (int k0 = 0; k0 < KDIM; k0 += 32)
      acc = wmma_bf16(ldfrag_g(arow + k0), ldfrag_g(brow + k0), acc);
    const int e    = n0 + col - NOFF;          // mask element index
    const float bs = b_mod[n0 + col - NOFF];
    #pragma unroll
    for (int r = 0; r < 8; ++r) {
      int mr = r + 8 * khal;
      s_mask[mr * KDIM + e] = f2bf(acc[r] + bs);
    }
  }
  __syncthreads();

  // ---- Phase 1b: offset channels + immediate bilinear sampling ------------
  const int pixb = m0 >> 12;               // batch (shared by tile)
  const int pixh = (m0 >> 6) & 63;         // row   (shared by tile)
  const int pixw0 = m0 & 63;               // first column
  for (int it = 0; it < 9; ++it) {         // 72 tiles / 8 waves, uniform
    const int t  = it * 8 + wave;          // offset N-tile 0..71
    const int n0 = t * 16;
    v8f acc = {0.f, 0.f, 0.f, 0.f, 0.f, 0.f, 0.f, 0.f};
    const unsigned short* brow = Wcat + (size_t)(n0 + col) * KDIM + khal * 16;
    for (int k0 = 0; k0 < KDIM; k0 += 32)
      acc = wmma_bf16(ldfrag_g(arow + k0), ldfrag_g(brow + k0), acc);
    const float bs = b_off[n0 + col];
    #pragma unroll
    for (int r = 0; r < 8; ++r) {
      int mr = r + 8 * khal;
      float v = acc[r] + bs;
      v = fminf(fmaxf(v, -MAXOFF), MAXOFF);
      s_off[wave][mr * 16 + col] = v;
    }
    __syncthreads();   // uniform barrier: offsets visible across lanes

    // sample 16 pixels x 8 elements (elements e0..e0+7), 4 tasks per lane
    const int e0 = t * 8;
    #pragma unroll
    for (int q = 0; q < 4; ++q) {
      const int task = lane + 32 * q;      // 0..127
      const int p  = task >> 3;            // pixel 0..15
      const int j  = task & 7;             // element within tile
      const int e  = e0 + j;               // global element c*9+kk
      const int c  = e / KK;
      const int kk = e % KK;
      const int ki = kk / 3, kj = kk % 3;
      const float dy = s_off[wave][p * 16 + 2 * j];
      const float dx = s_off[wave][p * 16 + 2 * j + 1];
      const float mv = bf2f(s_mask[p * KDIM + e]);
      const float py = dy + (float)(pixh + ki - 1);
      const float px = dx + (float)(pixw0 + p + kj - 1);
      const bool valid = (py > -1.f) && (py < (float)Hdim) &&
                         (px > -1.f) && (px < (float)Wdim);
      const float y0f = floorf(py), x0f = floorf(px);
      const float wy = py - y0f,    wx = px - x0f;
      const int y0 = (int)y0f, x0i = (int)x0f;
      const float* xb = x + (((size_t)pixb * Cch + c) << 12);
      float g00 = 0.f, g01 = 0.f, g10 = 0.f, g11 = 0.f;
      {
        int yc = min(max(y0, 0), Hdim - 1);
        int xc = min(max(x0i, 0), Wdim - 1);
        int yd = min(max(y0 + 1, 0), Hdim - 1);
        int xd = min(max(x0i + 1, 0), Wdim - 1);
        bool iy0 = (y0 >= 0) && (y0 < Hdim);
        bool iy1 = (y0 + 1 >= 0) && (y0 + 1 < Hdim);
        bool ix0 = (x0i >= 0) && (x0i < Wdim);
        bool ix1 = (x0i + 1 >= 0) && (x0i + 1 < Wdim);
        if (iy0 && ix0) g00 = xb[yc * Wdim + xc];
        if (iy0 && ix1) g01 = xb[yc * Wdim + xd];
        if (iy1 && ix0) g10 = xb[yd * Wdim + xc];
        if (iy1 && ix1) g11 = xb[yd * Wdim + xd];
      }
      float val = (1.f - wy) * (1.f - wx) * g00 + (1.f - wy) * wx * g01 +
                  wy * (1.f - wx) * g10 + wy * wx * g11;
      float s = valid ? val * mv : 0.f;
      s_samp[p * KDIM + e] = f2bf(s);
    }
  }
  __syncthreads();

  // ---- Phase 3: D[16 x 128] = S[16 x 576] x WcT, one 16-wide N-tile/wave --
  {
    const int n0 = wave * 16;              // O channels n0..n0+15
    v8f acc = {0.f, 0.f, 0.f, 0.f, 0.f, 0.f, 0.f, 0.f};
    const unsigned short* srow = &s_samp[col * KDIM + khal * 16];
    const unsigned short* brow = Wc + (size_t)(n0 + col) * KDIM + khal * 16;
    for (int k0 = 0; k0 < KDIM; k0 += 32)
      acc = wmma_bf16(ldfrag_s(srow + k0), ldfrag_g(brow + k0), acc);
    const int o = n0 + col;
    #pragma unroll
    for (int r = 0; r < 8; ++r) {
      int p = r + 8 * khal;
      out[(((size_t)pixb * Och + o) << 12) + pixh * Wdim + (pixw0 + p)] = acc[r];
    }
  }
}

// ---------------------------------------------------------------------------
extern "C" void kernel_launch(void* const* d_in, const int* in_sizes, int n_in,
                              void* d_out, int out_size, void* d_ws, size_t ws_size,
                              hipStream_t stream) {
  const float* x      = (const float*)d_in[0];
  const float* w_off  = (const float*)d_in[1];
  const float* b_off  = (const float*)d_in[2];
  const float* w_mod  = (const float*)d_in[3];
  const float* b_mod  = (const float*)d_in[4];
  const float* w_conv = (const float*)d_in[5];
  float* out = (float*)d_out;

  unsigned short* A    = (unsigned short*)d_ws;               // 32768*576
  unsigned short* Wcat = A    + (size_t)MTOT * KDIM;          // 1728*576
  unsigned short* Wc   = Wcat + (size_t)NCAT * KDIM;          // 128*576

  {
    size_t n = (size_t)MTOT * KDIM;
    im2col_bf16<<<(unsigned)((n + 255) / 256), 256, 0, stream>>>(x, A);
  }
  {
    size_t n = (size_t)NCAT * KDIM + (size_t)Och * KDIM;
    packw_bf16<<<(unsigned)((n + 255) / 256), 256, 0, stream>>>(
        w_off, w_mod, w_conv, Wcat, Wc);
  }
  deform_fused<<<MTOT / 16, 256, 0, stream>>>(x, b_off, b_mod, A, Wcat, Wc, out);
}
